// GNN_1288490189621
// MI455X (gfx1250) — compile-verified, matrix operand order checked by reading, not run on hardware
//
#include <hip/hip_runtime.h>
#include <hip/hip_bf16.h>
#include <cstdint>

typedef __attribute__((ext_vector_type(16))) __bf16 v16bf;
typedef __attribute__((ext_vector_type(8)))  __bf16 v8bf;
typedef __attribute__((ext_vector_type(8)))  float  v8f;

#define N_NODES 50000
#define NP      50048      // N padded to 128 (391 * 128)
#define MAXP    2048       // max padded dim (2000 -> 2048)

static inline int pad64(int x) { return (x + 63) & ~63; }

// ---------------------------------------------------------------------------
// Weights: W [K, Dout] f32 (row-major) -> Wt [Dp, Kp] bf16 (transposed, 0-pad)
// ---------------------------------------------------------------------------
__global__ __launch_bounds__(256) void k_wt(const float* __restrict__ W,
                                            __bf16* __restrict__ Wt,
                                            int K, int Dout, int Kp, int Dp)
{
    size_t idx = (size_t)blockIdx.x * blockDim.x + threadIdx.x;
    size_t total = (size_t)Dp * Kp;
    if (idx >= total) return;
    int d = (int)(idx / Kp);
    int k = (int)(idx % Kp);
    float v = (d < Dout && k < K) ? W[(size_t)k * Dout + d] : 0.0f;
    Wt[idx] = (__bf16)v;
}

// ---------------------------------------------------------------------------
// H[n,k] = bf16( (1-s)*relu(accIn[n,k]) + s*tra[n,k] ), zero in padding
// ---------------------------------------------------------------------------
__global__ __launch_bounds__(256) void k_mix(const float* __restrict__ accIn,
                                             const float* __restrict__ tra,
                                             __bf16* __restrict__ H,
                                             int K, int Kp, int inW,
                                             float s, int do_relu)
{
    size_t idx = (size_t)blockIdx.x * blockDim.x + threadIdx.x;
    size_t total = (size_t)NP * Kp;
    if (idx >= total) return;
    int n = (int)(idx / Kp);
    int k = (int)(idx % Kp);
    float v = 0.0f;
    if (n < N_NODES && k < K) {
        float a = accIn[(size_t)n * inW + k];
        if (do_relu) a = fmaxf(a, 0.0f);
        v = tra ? ((1.0f - s) * a + s * tra[(size_t)n * K + k]) : a;
    }
    H[idx] = (__bf16)v;
}

// ---------------------------------------------------------------------------
// Load one 16-bf16 WMMA fragment: two contiguous 16-byte runs (K and K+16)
// ---------------------------------------------------------------------------
__device__ __forceinline__ v16bf ldfrag(const __bf16* p)
{
    v8bf x0 = *(const v8bf*)(p);
    v8bf x1 = *(const v8bf*)(p + 16);
    return __builtin_shufflevector(x0, x1, 0,1,2,3,4,5,6,7,8,9,10,11,12,13,14,15);
}

// ---------------------------------------------------------------------------
// GEMM: out[NP, Dp] f32 = H[NP, Kp] bf16 @ Wt[Dp, Kp]^T bf16
// Templated on KP so all fragment addresses are base + immediate offsets.
// One wave computes 16 rows x 64 cols (A fragment reused across 4 WMMAs).
// Block = 8 waves stacked on M -> 128 x 64 tile.
// Grid: x = N-strip (fast), y = M-strip (slow) so consecutive blocks reuse
// the same 128-row A tile out of L1/L2 -> H streamed from HBM only once.
// Software-pipelined: next K-slice fragments load before current WMMAs.
// ---------------------------------------------------------------------------
template <int KP>
__global__ __launch_bounds__(256) void k_gemm(const __bf16* __restrict__ H,
                                              const __bf16* __restrict__ Wt,
                                              float* __restrict__ out,
                                              int Dp)
{
    const int lane = threadIdx.x & 31;
    const int wv   = threadIdx.x >> 5;            // 0..7 -> M sub-tile
    const int half = lane >> 4;
    const int lr   = lane & 15;

    const int row0 = (blockIdx.y * 8 + wv) * 16;  // wave's 16-row strip (slow axis)
    const int col0 = blockIdx.x * 64;             // wave's 64-col strip (fast axis)

    const __bf16* aptr = H  + (size_t)(row0 + lr) * KP + half * 8;
    const __bf16* bptr = Wt + (size_t)(col0 + lr) * KP + half * 8;

    v8f acc[4] = {};

    // prologue: stage 0
    v16bf a_cur = ldfrag(aptr);
    v16bf b_cur[4];
#pragma unroll
    for (int j = 0; j < 4; ++j)
        b_cur[j] = ldfrag(bptr + j * 16 * KP);

    // steady state: load slice k0, compute slice k0-32
#pragma unroll 2
    for (int k0 = 32; k0 < KP; k0 += 32) {
        v16bf a_nxt = ldfrag(aptr + k0);
        __builtin_prefetch(aptr + k0 + 32, 0, 1);   // global_prefetch_b8
        v16bf b_nxt[4];
#pragma unroll
        for (int j = 0; j < 4; ++j)
            b_nxt[j] = ldfrag(bptr + j * 16 * KP + k0);

#pragma unroll
        for (int j = 0; j < 4; ++j)
            acc[j] = __builtin_amdgcn_wmma_f32_16x16x32_bf16(
                         false, a_cur, false, b_cur[j],
                         (short)0, acc[j], false, false);

        a_cur = a_nxt;
#pragma unroll
        for (int j = 0; j < 4; ++j) b_cur[j] = b_nxt[j];
    }

    // epilogue: last slice
#pragma unroll
    for (int j = 0; j < 4; ++j)
        acc[j] = __builtin_amdgcn_wmma_f32_16x16x32_bf16(
                     false, a_cur, false, b_cur[j],
                     (short)0, acc[j], false, false);

    // C/D layout: VGPR v <-> row (v + 8*half), col = lr (within each 16-tile)
#pragma unroll
    for (int j = 0; j < 4; ++j) {
        float* orow = out + (size_t)(row0 + 8 * half) * Dp + col0 + j * 16 + lr;
#pragma unroll
        for (int v = 0; v < 8; ++v)
            orow[(size_t)v * Dp] = acc[j][v];
    }
}

// ---------------------------------------------------------------------------
// SPMM: acc[rows[e], d] += vals[e] * sup[cols[e], d]
// Grid: (edge blocks, column chunks). cw = 1<<cwShift columns per edge,
// epb = 256>>cwShift edges per block. No integer division.
// ---------------------------------------------------------------------------
__global__ __launch_bounds__(256) void k_spmm(const int* __restrict__ rows,
                                              const int* __restrict__ cols,
                                              const float* __restrict__ vals,
                                              const float* __restrict__ sup,
                                              float* __restrict__ acc,
                                              long long E, int D, int Dp,
                                              int cwShift)
{
    const int t   = threadIdx.x;
    const int cw  = 1 << cwShift;
    const int epb = 256 >> cwShift;
    long long e = (long long)blockIdx.x * epb + (t >> cwShift);
    int       d = blockIdx.y * cw + (t & (cw - 1));
    if (e >= E || d >= D) return;
    int r = rows[e];
    int c = cols[e];
    float v = vals[e] * sup[(size_t)c * Dp + d];
    atomicAdd(&acc[(size_t)r * Dp + d], v);
}

// ---------------------------------------------------------------------------
// Strip padding: out[n, 0..9] = acc[n*Dp + 0..9]  (no ReLU on final layer)
// ---------------------------------------------------------------------------
__global__ __launch_bounds__(256) void k_copy(const float* __restrict__ acc,
                                              float* __restrict__ out, int Dp)
{
    int idx = blockIdx.x * blockDim.x + threadIdx.x;
    if (idx >= N_NODES * 10) return;
    int n = idx / 10;
    int d = idx - n * 10;
    out[idx] = acc[(size_t)n * Dp + d];
}

// ---------------------------------------------------------------------------
extern "C" void kernel_launch(void* const* d_in, const int* in_sizes, int n_in,
                              void* d_out, int out_size, void* d_ws, size_t ws_size,
                              hipStream_t stream)
{
    const float* x    = (const float*)d_in[0];
    const int*   rows = (const int*)  d_in[1];
    const int*   cols = (const int*)  d_in[2];
    const float* vals = (const float*)d_in[3];
    const float* tra[4] = { (const float*)d_in[4], (const float*)d_in[5],
                            (const float*)d_in[6], (const float*)d_in[7] };
    const float* W[5]   = { (const float*)d_in[8],  (const float*)d_in[9],
                            (const float*)d_in[10], (const float*)d_in[11],
                            (const float*)d_in[12] };
    const long long E = in_sizes[1];

    // Workspace carve-out (reused across all 5 layers via stream ordering)
    char* ws = (char*)d_ws;
    size_t off = 0;
    auto carve = [&](size_t bytes) -> void* {
        void* p = ws + off;
        off = (off + bytes + 255) & ~(size_t)255;
        return p;
    };
    __bf16* Hbf = (__bf16*)carve((size_t)NP * MAXP * sizeof(__bf16));     // gemm A input
    __bf16* Wt  = (__bf16*)carve((size_t)MAXP * MAXP * sizeof(__bf16));   // gemm B input
    float*  sup = (float*) carve((size_t)NP * MAXP * sizeof(float));      // gemm output
    float*  acc = (float*) carve((size_t)NP * MAXP * sizeof(float));      // spmm output

    const int Kdims[5] = { 512, 500, 500, 2000, 10 };
    const int Ddims[5] = { 500, 500, 2000, 10, 10 };
    const float sigma = 0.5f;

    for (int l = 0; l < 5; ++l) {
        const int K  = Kdims[l], D = Ddims[l];
        const int Kp = pad64(K), Dp = pad64(D);

        // 1) weight transpose+convert -> Wt [Dp, Kp] bf16
        {
            size_t total = (size_t)Dp * Kp;
            k_wt<<<(unsigned)((total + 255) / 256), 256, 0, stream>>>(W[l], Wt, K, D, Kp, Dp);
        }

        // 2) build bf16 GEMM input: l==0 -> x; else relu(prev acc) mixed with tra
        {
            const float* src = (l == 0) ? x : acc;
            const float* trp = (l == 0) ? nullptr : tra[l - 1];
            const int inW    = Kp;               // prev Dp == pad64(K) == Kp (512 for x)
            size_t total = (size_t)NP * Kp;
            k_mix<<<(unsigned)((total + 255) / 256), 256, 0, stream>>>(
                src, trp, Hbf, K, Kp, inW, sigma, (l == 0) ? 0 : 1);
        }

        // 3) dense GEMM via WMMA bf16 -> sup [NP, Dp] f32
        //    grid.x = N-strips (fast-varying) so consecutive blocks share A tile
        {
            dim3 grid(Dp / 64, NP / 128);
            if (Kp == 512)
                k_gemm<512><<<grid, 256, 0, stream>>>(Hbf, Wt, sup, Dp);
            else if (Kp == 2048)
                k_gemm<2048><<<grid, 256, 0, stream>>>(Hbf, Wt, sup, Dp);
            else
                k_gemm<64><<<grid, 256, 0, stream>>>(Hbf, Wt, sup, Dp);
        }

        // 4) zero accumulator, then SPMM gather + atomic scatter
        hipMemsetAsync(acc, 0, (size_t)NP * Dp * sizeof(float), stream);
        {
            const int cwShift = (D <= 16) ? 4 : (D <= 64) ? 6 : 8;
            const int cw  = 1 << cwShift;
            const int epb = 256 >> cwShift;
            dim3 grid((unsigned)((E + epb - 1) / epb), (unsigned)((D + cw - 1) / cw));
            k_spmm<<<grid, 256, 0, stream>>>(rows, cols, vals, sup, acc, E, D, Dp, cwShift);
        }
    }

    // 5) final output (no ReLU): [N, 10] from padded [NP, 64]
    k_copy<<<(N_NODES * 10 + 255) / 256, 256, 0, stream>>>(acc, (float*)d_out, 64);
}
